// ScaledCosineAttention_1683627180675
// MI455X (gfx1250) — matrix operand order repaired and checked by reading, not compiled, not run on hardware
//
#include <hip/hip_runtime.h>
#include <math.h>

// Scaled cosine attention for gfx1250 / MI455X, two-pass:
//   pass 1: L2-normalize Q,K -> f16; V -> f16 transposed via LDS (coalesced)
//   pass 2: flash attention with BOUNDED-score softmax (cosine attention:
//           |q.k| <= 1 so exp(scale*(s-1)) needs no max tracking), row sums
//           computed by a ones-matrix WMMA, async double-buffered K/V tiles
//           (ASYNCcnt), per-wave LDS P re-pack (DScnt), raw v_exp_f32.
// Workspace requirement: 3 * 4*8*2048*64 * 2B = 25,165,824 bytes.

typedef __attribute__((ext_vector_type(16))) _Float16 v16h;
typedef __attribute__((ext_vector_type(8)))  float    v8f;
typedef __attribute__((__vector_size__(16))) int      i32x4;

#define NB     4
#define SEQ    2048
#define NH     8
#define DH     64
#define DM     (NH * DH)          // 512
#define BQ     128                // queries per workgroup (8 waves x 16)
#define BK2    64                 // keys per buffered tile (2 x 32 sub-tiles)
#define NWAVE  8
#define NT     (SEQ / BK2)        // 32 key tiles
#define LOG_LOW  (-4.605170185988091f)   // ln(0.01)
#define LOG_HIGH ( 4.605170185988092f)   // ln(100)
#define LOG2E    (1.4426950408889634f)

#define AS1 __attribute__((address_space(1)))
#define AS3 __attribute__((address_space(3)))

#if defined(__has_builtin)
#if __has_builtin(__builtin_amdgcn_global_load_async_to_lds_b128)
#define HAVE_ASYNC_LDS 1
#endif
#if __has_builtin(__builtin_amdgcn_exp2f)
#define FAST_EXP2(x) __builtin_amdgcn_exp2f(x)   // raw v_exp_f32; tails flush to 0
#endif
#endif
#ifndef FAST_EXP2
#define FAST_EXP2(x) exp2f(x)
#endif

// copy 16 bytes global -> LDS (async on CDNA5, tracked by ASYNCcnt)
__device__ __forceinline__ void async_copy16(const _Float16* g, _Float16* l) {
#ifdef HAVE_ASYNC_LDS
    __builtin_amdgcn_global_load_async_to_lds_b128(
        (AS1 i32x4*)(uintptr_t)g,
        (AS3 i32x4*)(uint32_t)(uintptr_t)l, 0, 0);
#else
    *(float4*)l = *(const float4*)g;   // sync fallback
#endif
}

// ---------------- pass 1: normalize + pack ----------------
__launch_bounds__(256)
__global__ void preprocess_kernel(const float* __restrict__ Q,
                                  const float* __restrict__ K,
                                  const float* __restrict__ V,
                                  _Float16* __restrict__ Qh,   // [b][h][s][d]
                                  _Float16* __restrict__ Kh,   // [b][h][s][d]
                                  _Float16* __restrict__ Vt)   // [b][h][d][s]
{
    __shared__ __align__(16) _Float16 sT[DH][128 + 8];  // V transpose staging

    const int tid  = threadIdx.x;
    const int row  = tid >> 1;       // 0..127
    const int half = tid & 1;        // two lanes per row
    const int b = blockIdx.z, h = blockIdx.y;
    const int sr = blockIdx.x * 128 + row;
    const size_t gi = ((size_t)(b * SEQ + sr)) * DM + h * DH + half * 32;
    const size_t bh = (size_t)(b * NH + h);

    {   // Q: L2 normalize over d=64, store f16 row-major
        const float* src = Q + gi;
        float buf[32]; float ss = 0.f;
        #pragma unroll
        for (int j = 0; j < 32; ++j) { buf[j] = src[j]; ss += buf[j] * buf[j]; }
        ss += __shfl_xor(ss, 1, 32);
        const float inv = 1.f / fmaxf(sqrtf(ss), 1e-12f);
        _Float16* dst = Qh + (bh * SEQ + sr) * DH + half * 32;
        #pragma unroll
        for (int j = 0; j < 32; ++j) dst[j] = (_Float16)(buf[j] * inv);
    }
    {   // K: same
        const float* src = K + gi;
        float buf[32]; float ss = 0.f;
        #pragma unroll
        for (int j = 0; j < 32; ++j) { buf[j] = src[j]; ss += buf[j] * buf[j]; }
        ss += __shfl_xor(ss, 1, 32);
        const float inv = 1.f / fmaxf(sqrtf(ss), 1e-12f);
        _Float16* dst = Kh + (bh * SEQ + sr) * DH + half * 32;
        #pragma unroll
        for (int j = 0; j < 32; ++j) dst[j] = (_Float16)(buf[j] * inv);
    }
    {   // V: f16 transpose through LDS, then coalesced b128 stores along s
        const float* src = V + gi;
        #pragma unroll
        for (int j = 0; j < 32; ++j)
            sT[half * 32 + j][row] = (_Float16)src[j];
        __syncthreads();
        const int d  = tid >> 2;     // 0..63
        const int sp = tid & 3;      // 32-half chunk along s
        _Float16* dstv = Vt + (bh * DH + d) * SEQ + blockIdx.x * 128 + sp * 32;
        const _Float16* srow = &sT[d][sp * 32];
        #pragma unroll
        for (int j = 0; j < 32; ++j) dstv[j] = srow[j];
    }
}

// ---------------- pass 2: flash attention ----------------
__launch_bounds__(256)
__global__ void cos_attn_kernel(const _Float16* __restrict__ Qh,
                                const _Float16* __restrict__ Kh,
                                const _Float16* __restrict__ Vt,
                                const float* __restrict__ LS,
                                float* __restrict__ Out)
{
    __shared__ __align__(16) _Float16 sK[2][BK2][DH + 8];    // k tile  [key][d]
    __shared__ __align__(16) _Float16 sV[2][DH][BK2 + 8];    // v tile  [d][key]
    __shared__ __align__(16) _Float16 sP[NWAVE][16][32 + 4]; // per-wave P staging

    const int tid  = threadIdx.x;
    const int wave = tid >> 5;
    const int lane = tid & 31;
    const int ln   = lane & 15;   // fragment row (A) / col (B,D) select
    const int hi   = lane >> 4;   // lane half

    const int b  = blockIdx.z;
    const int h  = blockIdx.y;
    const int qt = blockIdx.x;
    const size_t bh = (size_t)(b * NH + h);

    const float scale = __expf(fminf(fmaxf(LS[h], LOG_LOW), LOG_HIGH));
    const float c1 = scale * LOG2E;     // p = exp2(c1*s - c1), s in [-1,1]

    const _Float16* Kbase = Kh + bh * SEQ * DH;
    const _Float16* Vbase = Vt + bh * DH * SEQ;

    // async tile copy: 64x64 halves per operand = 8KB; 4 x b128 per thread
    auto issue_tile = [&](int kt, int buf) {
        #pragma unroll
        for (int i = 0; i < 2; ++i) {
            const int idx  = tid + 256 * i;
            const int row  = idx >> 3;      // 0..63
            const int part = idx & 7;       // 8-half (16B) chunk
            async_copy16(Kbase + ((size_t)(kt * BK2 + row)) * DH + part * 8,
                         &sK[buf][row][part * 8]);
            async_copy16(Vbase + (size_t)row * SEQ + kt * BK2 + part * 8,
                         &sV[buf][row][part * 8]);
        }
    };

    // ---- Q A-fragments straight from global f16 ----
    // A-layout (ISA 16-bit A 16x32): lane l: M=l&15; elem e -> K =
    // (e<8?0:16) + (l>>4)*8 + (e&7).  Same K packing used for matching B.
    const int growq = qt * BQ + wave * 16 + ln;
    const _Float16* qp = Qh + (bh * SEQ + growq) * DH;
    v16h qa0, qa1, onesb;
    #pragma unroll
    for (int e = 0; e < 16; ++e) {
        const int kk = ((e < 8) ? 0 : 16) + hi * 8 + (e & 7);
        qa0[e] = qp[kk];
        qa1[e] = qp[32 + kk];
        onesb[e] = (_Float16)1.0f;       // B of ones -> row-sum WMMA
    }

    v8f o0 = {}, o1 = {}, o2 = {}, o3 = {};   // output accum (16 x 64)
    v8f lacc = {};                            // softmax denominator accum

    issue_tile(0, 0);   // prologue

    for (int kt = 0; kt < NT; ++kt) {
        const int cur = kt & 1;
        if (kt + 1 < NT) {
            issue_tile(kt + 1, cur ^ 1);                      // prefetch next
            asm volatile("s_wait_asynccnt 0x4" ::: "memory"); // tile kt done
        } else {
            asm volatile("s_wait_asynccnt 0x0" ::: "memory");
        }
        __syncthreads();   // all waves' copies of tile kt visible

        #pragma unroll
        for (int sub = 0; sub < 2; ++sub) {       // two 32-key sub-tiles
            const int kb = sub * 32;

            // ---- S = Q * K^T (16 queries x 32 keys) ----
            v8f s0c = {}, s1c = {};
            #pragma unroll
            for (int c = 0; c < 2; ++c) {
                v16h kb0, kb1;
                #pragma unroll
                for (int e = 0; e < 16; ++e) {
                    const int kr = ((e < 8) ? 0 : 16) + hi * 8 + (e & 7); // d inside 32-chunk
                    kb0[e] = sK[cur][kb + ln     ][c * 32 + kr];
                    kb1[e] = sK[cur][kb + 16 + ln][c * 32 + kr];
                }
                const v16h qa = (c == 0) ? qa0 : qa1;
                s0c = __builtin_amdgcn_wmma_f32_16x16x32_f16(false, qa, false, kb0,
                                                             (short)0, s0c, false, false);
                s1c = __builtin_amdgcn_wmma_f32_16x16x32_f16(false, qa, false, kb1,
                                                             (short)0, s1c, false, false);
            }

            // ---- bounded softmax: p = exp(scale*s - scale), no max pass ----
            // store P straight to per-wave LDS in D-layout (row = r+8*hi)
            #pragma unroll
            for (int r = 0; r < 8; ++r) {
                const float p0 = FAST_EXP2(fmaf(s0c[r], c1, -c1));
                const float p1 = FAST_EXP2(fmaf(s1c[r], c1, -c1));
                sP[wave][r + 8 * hi][ln]      = (_Float16)p0;
                sP[wave][r + 8 * hi][16 + ln] = (_Float16)p1;
            }
            asm volatile("s_wait_dscnt 0" ::: "memory"); // same-wave DS in-order
            v16h pa;
            #pragma unroll
            for (int e = 0; e < 16; ++e) {
                const int kk = ((e < 8) ? 0 : 16) + hi * 8 + (e & 7);
                pa[e] = sP[wave][ln][kk];
            }

            // ---- denominator: row sums via ones-matrix WMMA ----
            lacc = __builtin_amdgcn_wmma_f32_16x16x32_f16(false, pa, false, onesb,
                                                          (short)0, lacc, false, false);

            // ---- O += P * V  (4 output 16-col groups over d) ----
            v16h vb0, vb1, vb2, vb3;
            #pragma unroll
            for (int e = 0; e < 16; ++e) {
                const int kr = kb + ((e < 8) ? 0 : 16) + hi * 8 + (e & 7);   // key index (B K-row)
                vb0[e] = sV[cur][ 0 + ln][kr];
                vb1[e] = sV[cur][16 + ln][kr];
                vb2[e] = sV[cur][32 + ln][kr];
                vb3[e] = sV[cur][48 + ln][kr];
            }
            o0 = __builtin_amdgcn_wmma_f32_16x16x32_f16(false, pa, false, vb0,
                                                        (short)0, o0, false, false);
            o1 = __builtin_amdgcn_wmma_f32_16x16x32_f16(false, pa, false, vb1,
                                                        (short)0, o1, false, false);
            o2 = __builtin_amdgcn_wmma_f32_16x16x32_f16(false, pa, false, vb2,
                                                        (short)0, o2, false, false);
            o3 = __builtin_amdgcn_wmma_f32_16x16x32_f16(false, pa, false, vb3,
                                                        (short)0, o3, false, false);
        }

        __syncthreads();   // everyone done reading tile kt before re-fill
    }

    // ---- finalize: divide by softmax denominator, write out ----
    #pragma unroll
    for (int r = 0; r < 8; ++r) {
        const float invl = 1.0f / lacc[r];
        const int grow = qt * BQ + wave * 16 + r + 8 * hi;
        float* dst = Out + ((size_t)(b * SEQ + grow)) * DM + h * DH;
        dst[ 0 + ln] = o0[r] * invl;
        dst[16 + ln] = o1[r] * invl;
        dst[32 + ln] = o2[r] * invl;
        dst[48 + ln] = o3[r] * invl;
    }
}

extern "C" void kernel_launch(void* const* d_in, const int* in_sizes, int n_in,
                              void* d_out, int out_size, void* d_ws, size_t ws_size,
                              hipStream_t stream) {
    (void)in_sizes; (void)n_in; (void)out_size; (void)ws_size;
    const float* q  = (const float*)d_in[0];
    const float* k  = (const float*)d_in[1];
    const float* v  = (const float*)d_in[2];
    const float* ls = (const float*)d_in[3];
    float* out = (float*)d_out;

    const size_t per = (size_t)NB * NH * SEQ * DH;   // 4,194,304 elems
    _Float16* Qh = (_Float16*)d_ws;                  // needs 3*per*2 = 24 MiB
    _Float16* Kh = Qh + per;
    _Float16* Vt = Kh + per;

    dim3 g1(SEQ / 128, NH, NB);
    preprocess_kernel<<<g1, 256, 0, stream>>>(q, k, v, Qh, Kh, Vt);

    dim3 g2(SEQ / BQ, NH, NB);
    cos_attn_kernel<<<g2, 256, 0, stream>>>(Qh, Kh, Vt, ls, out);
}
